// MambaEncoder_48541720379858
// MI455X (gfx1250) — matrix operand order repaired and checked by reading, not compiled, not run on hardware
//
#include <hip/hip_runtime.h>
#include <cstddef>

// ---------------- types ----------------
typedef __attribute__((ext_vector_type(16))) _Float16 v16h;
typedef __attribute__((ext_vector_type(8)))  _Float16 v8h;
typedef __attribute__((ext_vector_type(8)))  float    v8f;

#define EPS 1e-6f

static __device__ __forceinline__ float silu_f(float x) {
    return x / (1.0f + __expf(-x));
}
static __device__ __forceinline__ float softplus_f(float x) {
    return (x > 20.0f) ? x : log1pf(__expf(x));
}

// ---------------- generic direct conv (stem 7x7 + down 3x3) ----------------
__global__ void conv2d_kernel(const float* __restrict__ x, const float* __restrict__ w,
                              const float* __restrict__ bias, float* __restrict__ out,
                              int B, int Cin, int H, int W, int Cout, int OH, int OW,
                              int KH, int KW, int stride, int pad) {
    int tid = blockIdx.x * blockDim.x + threadIdx.x;
    int total = B * Cout * OH * OW;
    if (tid >= total) return;
    int ox = tid % OW; int t = tid / OW;
    int oy = t % OH;  t /= OH;
    int oc = t % Cout; int b = t / Cout;
    float acc = bias[oc];
    for (int ic = 0; ic < Cin; ++ic) {
        const float* xb = x + ((size_t)(b * Cin + ic) * H) * W;
        const float* wb = w + ((size_t)(oc * Cin + ic) * KH) * KW;
        for (int ky = 0; ky < KH; ++ky) {
            int iy = oy * stride - pad + ky;
            if (iy < 0 || iy >= H) continue;
            for (int kx = 0; kx < KW; ++kx) {
                int ix = ox * stride - pad + kx;
                if (ix < 0 || ix >= W) continue;
                acc += xb[iy * W + ix] * wb[ky * KW + kx];
            }
        }
    }
    out[tid] = acc;
}

// ---------------- channels-first layernorm (per-pixel over C) ----------------
__global__ void ln_cf_kernel(const float* __restrict__ x, float* __restrict__ y,
                             const float* __restrict__ w, const float* __restrict__ b,
                             int B, int C, int HW) {
    int tid = blockIdx.x * blockDim.x + threadIdx.x;
    if (tid >= B * HW) return;
    int p = tid % HW; int bb = tid / HW;
    const float* xp = x + (size_t)bb * C * HW + p;
    float* yp = y + (size_t)bb * C * HW + p;
    float mu = 0.f;
    for (int c = 0; c < C; ++c) mu += xp[(size_t)c * HW];
    mu /= (float)C;
    float var = 0.f;
    for (int c = 0; c < C; ++c) { float d = xp[(size_t)c * HW] - mu; var += d * d; }
    var /= (float)C;
    float inv = rsqrtf(var + EPS);
    for (int c = 0; c < C; ++c)
        yp[(size_t)c * HW] = (xp[(size_t)c * HW] - mu) * inv * w[c] + b[c];
}

// ---------------- last-dim layernorm over contiguous rows ----------------
__global__ void ln_rows_kernel(const float* __restrict__ x, float* __restrict__ y,
                               const float* __restrict__ w, const float* __restrict__ b,
                               int rows, int D) {
    int r = blockIdx.x * blockDim.x + threadIdx.x;
    if (r >= rows) return;
    const float* xp = x + (size_t)r * D;
    float* yp = y + (size_t)r * D;
    float mu = 0.f;
    for (int c = 0; c < D; ++c) mu += xp[c];
    mu /= (float)D;
    float var = 0.f;
    for (int c = 0; c < D; ++c) { float d = xp[c] - mu; var += d * d; }
    var /= (float)D;
    float inv = rsqrtf(var + EPS);
    for (int c = 0; c < D; ++c) yp[c] = (xp[c] - mu) * inv * w[c] + b[c];
}

// ---------------- layout shuffles ----------------
__global__ void nchw_to_blc_kernel(const float* __restrict__ x, float* __restrict__ y,
                                   int B, int C, int L) {
    int tid = blockIdx.x * blockDim.x + threadIdx.x;
    if (tid >= B * C * L) return;
    int c = tid % C; int t = tid / C;
    int p = t % L;  int b = t / L;
    y[tid] = x[((size_t)(b * C + c)) * L + p];
}
__global__ void blc_to_nchw_kernel(const float* __restrict__ x, float* __restrict__ y,
                                   int B, int C, int L) {
    int tid = blockIdx.x * blockDim.x + threadIdx.x;
    if (tid >= B * C * L) return;
    int p = tid % L; int t = tid / L;
    int c = t % C;  int b = t / C;
    y[tid] = x[((size_t)(b * L + p)) * C + c];
}

// ---------------- f32 -> f16 packing (zero-pad K to Kp; rows as-is) ----------
__global__ void pack_a_f16_kernel(const float* __restrict__ src, _Float16* __restrict__ dst,
                                  int M, int K, int lda, int Kp) {
    int tid = blockIdx.x * blockDim.x + threadIdx.x;
    if (tid >= M * Kp) return;
    int k = tid % Kp; int m = tid / Kp;
    dst[tid] = (k < K) ? (_Float16)src[(size_t)m * lda + k] : (_Float16)0.f;
}
// weights: also zero-pad rows N -> Np so GEMM loads are guard-free
__global__ void pack_w_f16_kernel(const float* __restrict__ src, _Float16* __restrict__ dst,
                                  int N, int K, int ldw, int Np, int Kp) {
    int tid = blockIdx.x * blockDim.x + threadIdx.x;
    if (tid >= Np * Kp) return;
    int k = tid % Kp; int n = tid / Kp;
    dst[tid] = (n < N && k < K) ? (_Float16)src[(size_t)n * ldw + k] : (_Float16)0.f;
}

// ---------------- WMMA GEMM: out[M,N] = Ah[M,Kp] * Wh[Np,Kp]^T ---------------
// f16 inputs pre-packed (K zero-padded to Kp%32==0, W rows padded to Np%32==0).
// One wave computes a 32x32 output tile: 4 WMMAs per 32-wide K chunk, all loads
// are aligned global_load_b128 with no guards. act: 0=none, 1=silu, 2=softplus.
__global__ __launch_bounds__(32)
void gemm_wmma_kernel(const _Float16* __restrict__ Ah, const _Float16* __restrict__ Wh,
                      const float* __restrict__ bias, float* __restrict__ out,
                      int M, int N, int Kp, int act) {
    const int lane = threadIdx.x;
    const int lm = lane & 15;
    const int hi = lane >> 4;
    const int m0 = blockIdx.y * 32;
    const int n0 = blockIdx.x * 32;

    const _Float16* a0p = Ah + (size_t)(m0 + lm) * Kp;
    const _Float16* a1p = Ah + (size_t)(m0 + 16 + lm) * Kp;
    const _Float16* b0p = Wh + (size_t)(n0 + lm) * Kp;
    const _Float16* b1p = Wh + (size_t)(n0 + 16 + lm) * Kp;

    v8f acc00 = {}, acc01 = {}, acc10 = {}, acc11 = {};

    for (int k0 = 0; k0 < Kp; k0 += 32) {
        // A 16x32 f16 fragment: lane holds two contiguous 8-half runs
        //   af[0..7]  = K[k0 + hi*16 .. +7],  af[8..15] = K[k0+hi*16+8 .. +7]
        v8h a0lo = *(const v8h*)(a0p + k0 + (hi << 4));
        v8h a0hi = *(const v8h*)(a0p + k0 + 8 + (hi << 4));
        v8h a1lo = *(const v8h*)(a1p + k0 + (hi << 4));
        v8h a1hi = *(const v8h*)(a1p + k0 + 8 + (hi << 4));
        v16h af0 = __builtin_shufflevector(a0lo, a0hi, 0,1,2,3,4,5,6,7,8,9,10,11,12,13,14,15);
        v16h af1 = __builtin_shufflevector(a1lo, a1hi, 0,1,2,3,4,5,6,7,8,9,10,11,12,13,14,15);
        // B 32x16 f16 fragment: lane holds one contiguous 16-half run K[k0+hi*16 .. +15]
        v16h bf0 = *(const v16h*)(b0p + k0 + (hi << 4));
        v16h bf1 = *(const v16h*)(b1p + k0 + (hi << 4));

        acc00 = __builtin_amdgcn_wmma_f32_16x16x32_f16(false, af0, false, bf0, (short)0, acc00, false, false);
        acc01 = __builtin_amdgcn_wmma_f32_16x16x32_f16(false, af0, false, bf1, (short)0, acc01, false, false);
        acc10 = __builtin_amdgcn_wmma_f32_16x16x32_f16(false, af1, false, bf0, (short)0, acc10, false, false);
        acc11 = __builtin_amdgcn_wmma_f32_16x16x32_f16(false, af1, false, bf1, (short)0, acc11, false, false);
    }

    // C/D f32 16x16: lanes0-15 -> N=lane, M=v ; lanes16-31 -> N=lane-16, M=8+v
    v8f accs[2][2] = {{acc00, acc01}, {acc10, acc11}};
#pragma unroll
    for (int tm = 0; tm < 2; ++tm) {
#pragma unroll
        for (int tn = 0; tn < 2; ++tn) {
            int ncol = n0 + tn * 16 + lm;
            if (ncol < N) {
                float bv = bias ? bias[ncol] : 0.f;
#pragma unroll
                for (int v = 0; v < 8; ++v) {
                    int mr = m0 + tm * 16 + (hi << 3) + v;
                    float val = accs[tm][tn][v] + bv;
                    if (act == 1) val = silu_f(val);
                    else if (act == 2) val = softplus_f(val);
                    out[(size_t)mr * N + ncol] = val;
                }
            }
        }
    }
}

// ---------------- causal depthwise conv1d (k=4, pad left 3) + SiLU ------------
__global__ void dwconv_silu_kernel(const float* __restrict__ xz,
                                   const float* __restrict__ w, const float* __restrict__ b,
                                   float* __restrict__ xc, int B, int L, int di) {
    int tid = blockIdx.x * blockDim.x + threadIdx.x;
    if (tid >= B * L * di) return;
    int c = tid % di; int t = tid / di;
    int l = t % L;   int bb = t / L;
    float acc = b[c];
#pragma unroll
    for (int j = 0; j < 4; ++j) {
        int lp = l + j - 3;
        if (lp >= 0)
            acc += w[c * 4 + j] * xz[((size_t)(bb * L + lp)) * (2 * di) + c];
    }
    xc[tid] = silu_f(acc);
}

// ---------------- selective scan ---------------------------------------------
// One 16-lane group per (b, d) channel; lane n = state index; wave32 holds 2 groups.
__global__ void scan_kernel(const float* __restrict__ u, const float* __restrict__ dt,
                            const float* __restrict__ A_log, const float* __restrict__ xdbl,
                            const float* __restrict__ Dp, float* __restrict__ y,
                            int B, int L, int di, int R, int dtr) {
    int tid = blockIdx.x * blockDim.x + threadIdx.x;
    int n = tid & 15;
    int g = tid >> 4;
    if (g >= B * di) return;          // uniform per-wave (B*di multiple of 32)
    int d = g % di, b = g / di;
    float a  = -__expf(A_log[d * 16 + n]);
    float Dd = Dp[d];
    float h = 0.f;
    for (int l = 0; l < L; ++l) {
        size_t base = (size_t)(b * L + l);
        float delta = dt[base * di + d];
        float uu    = u[base * di + d];
        float Bn    = xdbl[base * R + dtr + n];
        float Cn    = xdbl[base * R + dtr + 16 + n];
        h = __expf(delta * a) * h + delta * Bn * uu;
        float p = h * Cn;
        p += __shfl_xor(p, 1);
        p += __shfl_xor(p, 2);
        p += __shfl_xor(p, 4);
        p += __shfl_xor(p, 8);
        if (n == 0) y[base * di + d] = p + uu * Dd;
    }
}

// ---------------- y *= silu(z)  (z = xz[..., di:]) ---------------------------
__global__ void mul_silu_kernel(float* __restrict__ y, const float* __restrict__ xz,
                                int BL, int di) {
    int tid = blockIdx.x * blockDim.x + threadIdx.x;
    if (tid >= BL * di) return;
    int c = tid % di; int bl = tid / di;
    y[tid] *= silu_f(xz[(size_t)bl * (2 * di) + di + c]);
}

// ---------------- host orchestration -----------------------------------------
static inline int cdiv(int a, int b) { return (a + b - 1) / b; }
static inline int rup(int a, int b) { return cdiv(a, b) * b; }

struct MambaP {
    const float *A_log, *D, *conv_b, *conv_w, *dt_b, *dt_w,
                *in_proj, *norm_b, *norm_w, *out_proj, *x_proj;
};

// pack + gemm launcher: out[M,N] = A[M,K](lda) * W[N,K](ldw)^T
static void launch_gemm(hipStream_t s, _Float16* Ah, _Float16* Whh,
                        const float* A, int lda, const float* W, int ldw,
                        const float* bias, float* out, int M, int N, int K, int act) {
    int Kp = rup(K, 32), Np = rup(N, 32);
    pack_a_f16_kernel<<<cdiv(M * Kp, 256), 256, 0, s>>>(A, Ah, M, K, lda, Kp);
    pack_w_f16_kernel<<<cdiv(Np * Kp, 256), 256, 0, s>>>(W, Whh, N, K, ldw, Np, Kp);
    dim3 grid(Np / 32, M / 32);   // M is always a multiple of 32 here
    gemm_wmma_kernel<<<grid, 32, 0, s>>>(Ah, Whh, bias, out, M, N, Kp, act);
}

extern "C" void kernel_launch(void* const* d_in, const int* in_sizes, int n_in,
                              void* d_out, int out_size, void* d_ws, size_t ws_size,
                              hipStream_t stream) {
    (void)out_size; (void)ws_size;
    const int B = 8;

    // --- locate x vs params (robust to either flattening order) ---
    int xi = (in_sizes[0] == 8 * 3 * 128 * 128) ? 0 : (n_in - 1);
    int idx = (xi == 0) ? 1 : 0;
    const float* X = (const float*)d_in[xi];
    auto F = [&](int i) { return (const float*)d_in[i]; };

    // params leaves in jax-pytree (alphabetical) order:
    const float *down_cb[3], *down_cw[3], *down_lnb[3], *down_lnw[3];
    for (int i = 0; i < 3; ++i) {
        down_cb[i] = F(idx++); down_cw[i] = F(idx++);
        down_lnb[i] = F(idx++); down_lnw[i] = F(idx++);
    }
    const float *on_b[4], *on_w[4];
    for (int i = 0; i < 4; ++i) { on_b[i] = F(idx++); on_w[i] = F(idx++); }
    MambaP Lp[4][2];
    for (int i = 0; i < 4; ++i)
        for (int j = 0; j < 2; ++j) {
            MambaP& p = Lp[i][j];
            p.A_log = F(idx++); p.D = F(idx++);
            p.conv_b = F(idx++); p.conv_w = F(idx++);
            p.dt_b = F(idx++);  p.dt_w = F(idx++);
            p.in_proj = F(idx++);
            p.norm_b = F(idx++); p.norm_w = F(idx++);
            p.out_proj = F(idx++); p.x_proj = F(idx++);
        }
    const float *stem_cb = F(idx++), *stem_cw = F(idx++),
                *stem_lnb = F(idx++), *stem_lnw = F(idx++);

    // --- workspace layout ---
    float* ws = (float*)d_ws;
    size_t o = 0;
    float* h    = ws + o; o += 2097152;   // NCHW feature map
    float* h2   = ws + o; o += 2097152;   // conv scratch
    float* tok  = ws + o; o += 2097152;   // (B,L,C)
    float* tokn = ws + o; o += 2097152;   // normalized tokens
    float* xz   = ws + o; o += 8388608;   // (B,L,2*di)
    float* xc   = ws + o; o += 4194304;   // (B,L,di)
    float* xdbl = ws + o; o += 2097152;   // (B,L,R)
    float* dtb  = ws + o; o += 4194304;   // (B,L,di)
    float* ybuf = ws + o; o += 4194304;   // (B,L,di)
    float* yo   = ws + o; o += 2097152;   // (B,L,C)
    _Float16* Ah  = (_Float16*)(ws + o); o += 2097152;  // 4M halves packed A
    _Float16* Whh = (_Float16*)(ws + o); o += 1048576;  // 2M halves packed W

    const int DIMS[4] = {64, 128, 256, 512};
    const int STR[3]  = {2, 1, 2};

    // --- stem: 7x7 s2 p3 (3 -> 64), 128x128 -> 64x64; then channel-first LN ---
    int C = DIMS[0], Hh = 64, Wh = 64;
    {
        int total = B * C * Hh * Wh;
        conv2d_kernel<<<cdiv(total, 256), 256, 0, stream>>>(
            X, stem_cw, stem_cb, h, B, 3, 128, 128, C, Hh, Wh, 7, 7, 2, 3);
        ln_cf_kernel<<<cdiv(B * Hh * Wh, 256), 256, 0, stream>>>(
            h, h, stem_lnw, stem_lnb, B, C, Hh * Wh);
    }

    size_t out_off = 0;
    for (int st = 0; st < 4; ++st) {
        if (st > 0) {
            int Cin = DIMS[st - 1], Cout = DIMS[st], sds = STR[st - 1];
            ln_cf_kernel<<<cdiv(B * Hh * Wh, 256), 256, 0, stream>>>(
                h, h, down_lnw[st - 1], down_lnb[st - 1], B, Cin, Hh * Wh);
            int OH = (Hh + 2 - 3) / sds + 1;
            int OW = (Wh + 2 - 3) / sds + 1;
            int total = B * Cout * OH * OW;
            conv2d_kernel<<<cdiv(total, 256), 256, 0, stream>>>(
                h, down_cw[st - 1], down_cb[st - 1], h2,
                B, Cin, Hh, Wh, Cout, OH, OW, 3, 3, sds, 1);
            float* tmp = h; h = h2; h2 = tmp;
            C = Cout; Hh = OH; Wh = OW;
        }

        int L = Hh * Wh, BL = B * L;
        int di = 2 * C, dtr = (C + 15) / 16, R = dtr + 32;

        for (int j = 0; j < 2; ++j) {
            const MambaP& p = Lp[st][j];
            nchw_to_blc_kernel<<<cdiv(B * C * L, 256), 256, 0, stream>>>(h, tok, B, C, L);
            ln_rows_kernel<<<cdiv(BL, 256), 256, 0, stream>>>(
                tok, tokn, p.norm_w, p.norm_b, BL, C);
            // in_proj: (BL,C) x (2di,C)^T -> xz (BL,2di)
            launch_gemm(stream, Ah, Whh, tokn, C, p.in_proj, C, nullptr, xz, BL, 2 * di, C, 0);
            // depthwise causal conv + silu -> xc
            dwconv_silu_kernel<<<cdiv(BL * di, 256), 256, 0, stream>>>(
                xz, p.conv_w, p.conv_b, xc, B, L, di);
            // x_proj: (BL,di) x (R,di)^T -> xdbl (BL,R)
            launch_gemm(stream, Ah, Whh, xc, di, p.x_proj, di, nullptr, xdbl, BL, R, di, 0);
            // dt: softplus(xdbl[:, :dtr] x dt_w^T + dt_b) -> (BL,di)
            launch_gemm(stream, Ah, Whh, xdbl, R, p.dt_w, dtr, p.dt_b, dtb, BL, di, dtr, 2);
            // selective scan -> ybuf
            scan_kernel<<<cdiv(B * di * 16, 256), 256, 0, stream>>>(
                xc, dtb, p.A_log, xdbl, p.D, ybuf, B, L, di, R, dtr);
            // gate with silu(z)
            mul_silu_kernel<<<cdiv(BL * di, 256), 256, 0, stream>>>(ybuf, xz, BL, di);
            // out_proj: (BL,di) x (C,di)^T -> yo (BL,C)
            launch_gemm(stream, Ah, Whh, ybuf, di, p.out_proj, di, nullptr, yo, BL, C, di, 0);
            blc_to_nchw_kernel<<<cdiv(B * C * L, 256), 256, 0, stream>>>(yo, h, B, C, L);
        }

        // stage output: channel-first LN written straight into d_out slice
        float* outp = (float*)d_out + out_off;
        ln_cf_kernel<<<cdiv(B * Hh * Wh, 256), 256, 0, stream>>>(
            h, outp, on_w[st], on_b[st], B, C, Hh * Wh);
        out_off += (size_t)B * C * Hh * Wh;
    }
}
